// SA_Layer_16174846837200
// MI455X (gfx1250) — matrix operand order repaired and checked by reading, not compiled, not run on hardware
//
#include <hip/hip_runtime.h>
#include <hip/hip_bf16.h>

#define DEVI __device__ __forceinline__

typedef __attribute__((ext_vector_type(16))) _Float16 v16h;
typedef __attribute__((ext_vector_type(8)))  float    v8f;

constexpr int B_ = 4, C_ = 128, N_ = 4096, I_ = 64;
constexpr int CN_ = C_ * N_;                 // 524288 elems per batch
constexpr int SEGS = 32, SEG_ELEMS = CN_ / SEGS;
constexpr int NT_ = N_ / 16;                 // 256 16-wide tiles along N

// ---------------------------------------------------------------------------
// WMMA fragment loaders (CDNA5 ISA §7.12.2, wave32).
// A (16x32 f16): row = lane&15; lane<16 K in {0..7,16..23}, lane>=16 K in
//   {8..15,24..31} -> two contiguous 8-element K-runs when s_k == 1.
// B (32x16 f16): col = lane&15; K split as for A -> two contiguous 8-element
//   K-runs when s_k == 1 (store B as N x K!).
// C/D (16x16 f32): col = lane&15, row = r + 8*(lane>>4).
// ---------------------------------------------------------------------------
DEVI v16h frag_a_h(const _Float16* p, int s_row, int s_k, int lane) {
  int row = lane & 15, grp = lane >> 4;
  v16h a;
#pragma unroll
  for (int vg = 0; vg < 8; ++vg)
#pragma unroll
    for (int q = 0; q < 2; ++q) {
      int k = ((vg < 4) ? 0 : 16) + grp * 8 + (vg & 3) * 2 + q;
      a[vg * 2 + q] = p[row * s_row + k * s_k];
    }
  return a;
}
DEVI v16h frag_b_h(const _Float16* p, int s_k, int s_n, int lane) {
  int n = lane & 15, grp = lane >> 4;
  v16h b;
#pragma unroll
  for (int v = 0; v < 8; ++v)
#pragma unroll
    for (int q = 0; q < 2; ++q) {
      int k = ((v < 4) ? 0 : 16) + grp * 8 + (v & 3) * 2 + q;
      b[v * 2 + q] = p[k * s_k + n * s_n];
    }
  return b;
}
DEVI v8f wmma16(v16h a, v16h b, v8f c) {
  return __builtin_amdgcn_wmma_f32_16x16x32_f16(false, a, false, b,
                                                (short)0, c, false, false);
}

// K0: one-shot f32 -> f16 conversion (weights); removes per-tile cvt chatter.
__global__ __launch_bounds__(256) void k_w2h(const float* __restrict__ src,
                                             _Float16* __restrict__ dst, int n) {
  int i = blockIdx.x * 256 + threadIdx.x;
  if (i < n) dst[i] = (_Float16)src[i];
}

// ---------------------------------------------------------------------------
// K1/K11: partial (sum, sumsq) per segment (deterministic tree reduction).
// ---------------------------------------------------------------------------
__global__ __launch_bounds__(256) void k_stats(const float* __restrict__ src,
                                               float* __restrict__ part) {
  int b = blockIdx.x / SEGS, sg = blockIdx.x % SEGS;
  const float* p = src + (size_t)b * CN_ + (size_t)sg * SEG_ELEMS;
  float s = 0.f, s2 = 0.f;
  for (int i = threadIdx.x; i < SEG_ELEMS; i += 256) {
    float v = p[i]; s += v; s2 += v * v;
  }
  __shared__ float sh[256], sh2[256];
  sh[threadIdx.x] = s; sh2[threadIdx.x] = s2;
  __syncthreads();
  for (int st = 128; st > 0; st >>= 1) {
    if (threadIdx.x < st) {
      sh[threadIdx.x]  += sh[threadIdx.x + st];
      sh2[threadIdx.x] += sh2[threadIdx.x + st];
    }
    __syncthreads();
  }
  if (threadIdx.x == 0) {
    part[blockIdx.x * 2 + 0] = sh[0];
    part[blockIdx.x * 2 + 1] = sh2[0];
  }
}

// K2/K12: finalize mean/rstd, emit per-(b,c) scale/shift.
__global__ __launch_bounds__(128) void k_gn_final(const float* __restrict__ part,
                                                  const float* __restrict__ g,
                                                  const float* __restrict__ beta,
                                                  float* __restrict__ scale,
                                                  float* __restrict__ shift) {
  int b = blockIdx.x;
  __shared__ float mean_s, rstd_s;
  if (threadIdx.x == 0) {
    float s = 0.f, s2 = 0.f;
    for (int i = 0; i < SEGS; ++i) {
      s  += part[(b * SEGS + i) * 2 + 0];
      s2 += part[(b * SEGS + i) * 2 + 1];
    }
    float m = s / (float)CN_;
    float var = s2 / (float)CN_ - m * m;
    mean_s = m; rstd_s = rsqrtf(var + 1e-5f);
  }
  __syncthreads();
  int c = threadIdx.x;
  float sc = rstd_s * g[c];
  scale[b * C_ + c] = sc;
  shift[b * C_ + c] = beta[c] - mean_s * sc;
}

// ---------------------------------------------------------------------------
// K3: fused normalize + transpose: xn_t[b][n][c] = (f16)(x[b][c][n]*sc+sh).
// LDS 32x32 tile transpose; coalesced along n on read, along c on write.
// ---------------------------------------------------------------------------
__global__ __launch_bounds__(256) void k_norm_t(const float* __restrict__ x,
                                                const float* __restrict__ scale,
                                                const float* __restrict__ shift,
                                                _Float16* __restrict__ xn_t) {
  __shared__ float tile[32][33];
  int b  = blockIdx.x >> 9;            // 512 blocks per batch
  int r  = blockIdx.x & 511;
  int ct = r >> 7;                     // c tile (0..3)
  int nt = r & 127;                    // n tile (0..127)
  int tx = threadIdx.x & 31, ty = threadIdx.x >> 5;
  int c0 = ct * 32, n0 = nt * 32;
  const float* xb = x + (size_t)b * CN_;
#pragma unroll
  for (int i = 0; i < 4; ++i) {
    int cl = ty + i * 8;
    int c = c0 + cl;
    tile[cl][tx] = xb[(size_t)c * N_ + n0 + tx] * scale[b * C_ + c]
                 + shift[b * C_ + c];
  }
  __syncthreads();
  _Float16* ot = xn_t + ((size_t)b * N_ + n0) * C_ + c0;
#pragma unroll
  for (int i = 0; i < 4; ++i) {
    int nl = ty + i * 8;
    ot[(size_t)nl * C_ + tx] = (_Float16)tile[tx][nl];
  }
}

// ---------------------------------------------------------------------------
// K4: all projections: [384x128] x [128xN] per batch via WMMA (f16 weights).
// Outputs token-major (contiguous epilogue stores) except v (kept [i][n],
// which is exactly K-contiguous for the later PV A-fragments).
// ---------------------------------------------------------------------------
__global__ __launch_bounds__(256) void k_project(
    const _Float16* __restrict__ xn_t,
    const _Float16* __restrict__ gq_wh, const float* __restrict__ gq_b,
    const _Float16* __restrict__ gkv_wh, const float* __restrict__ gkv_b,
    const _Float16* __restrict__ lqkv_wh, const float* __restrict__ lqkv_b,
    const float* __restrict__ dw_w, const float* __restrict__ dw_b,
    _Float16* __restrict__ qc_t, _Float16* __restrict__ k_t,
    _Float16* __restrict__ v_cm,
    _Float16* __restrict__ lq_t, _Float16* __restrict__ lk_t,
    float* __restrict__ lv_t) {
  int ng = blockIdx.x & 31;
  int t  = blockIdx.x >> 5;
  int ot = t % 24, b = t / 24;
  int lane = threadIdx.x & 31, w = threadIdx.x >> 5;
  int n0 = (ng * 8 + w) * 16;
  const _Float16* xb = xn_t + ((size_t)b * N_) * C_;
  const _Float16* W; int orow0;
  if (ot < 4)       { W = gq_wh;   orow0 = ot * 16; }
  else if (ot < 12) { W = gkv_wh;  orow0 = (ot - 4) * 16; }
  else              { W = lqkv_wh; orow0 = (ot - 12) * 16; }
  v8f d = {};
#pragma unroll
  for (int kk = 0; kk < 4; ++kk) {
    v16h a  = frag_a_h(W + (size_t)orow0 * C_ + kk * 32, C_, 1, lane);
    v16h bb = frag_b_h(xb + (size_t)n0 * C_ + kk * 32, 1, C_, lane);
    d = wmma16(a, bb, d);
  }
  int col = lane & 15, grp = lane >> 4;
  int n = n0 + col;
#pragma unroll
  for (int r = 0; r < 8; ++r) {
    int orow = orow0 + r + 8 * grp;
    float val = d[r];
    if (ot < 4) {
      val += gq_b[orow];
      qc_t[((size_t)b * N_ + n) * I_ + orow] = (_Float16)val;   // contiguous
    } else if (ot < 12) {
      val += gkv_b[orow];
      if (orow < 64)
        k_t[((size_t)b * N_ + n) * I_ + orow] = (_Float16)val;  // contiguous
      else
        v_cm[((size_t)b * I_ + orow - 64) * N_ + n] = (_Float16)val;
    } else {
      val += lqkv_b[orow];
      int ch = orow & 63;
      val = val * dw_w[ch] + dw_b[ch];
      size_t ix = ((size_t)b * N_ + n) * I_ + ch;               // contiguous
      if (orow < 64)       lq_t[ix] = (_Float16)val;
      else if (orow < 128) lk_t[ix] = (_Float16)val;
      else                 lv_t[ix] = val;
    }
  }
}

// K5: a = (f16)(lq * lk) elementwise (token-major layouts match).
__global__ __launch_bounds__(256) void k_amul(const _Float16* __restrict__ lq,
                                              const _Float16* __restrict__ lk,
                                              _Float16* __restrict__ a_t) {
  size_t idx = (size_t)blockIdx.x * 256 + threadIdx.x;
  if (idx < (size_t)B_ * I_ * N_)
    a_t[idx] = (_Float16)((float)lq[idx] * (float)lk[idx]);
}

// K6/K7: fc GEMM (64x64, K=64): MODE 0 -> SiLU -> s1_t; MODE 1 -> tanh*lv
// into y_t channels 0..63. All operands/outputs token-major.
template <int MODE>
__global__ __launch_bounds__(256) void k_fc(const _Float16* __restrict__ fc_wh,
                                            const float* __restrict__ fc_b,
                                            const _Float16* __restrict__ in_t,
                                            const float* __restrict__ lv_t,
                                            _Float16* __restrict__ out_t) {
  int ng = blockIdx.x & 31;
  int t  = blockIdx.x >> 5;
  int ot = t & 3, b = t >> 2;
  int lane = threadIdx.x & 31, w = threadIdx.x >> 5;
  int n0 = (ng * 8 + w) * 16;
  const _Float16* ib = in_t + ((size_t)b * N_) * I_;
  v8f d = {};
#pragma unroll
  for (int kk = 0; kk < 2; ++kk) {
    v16h a  = frag_a_h(fc_wh + (size_t)ot * 16 * I_ + kk * 32, I_, 1, lane);
    v16h bb = frag_b_h(ib + (size_t)n0 * I_ + kk * 32, 1, I_, lane);
    d = wmma16(a, bb, d);
  }
  int col = lane & 15, grp = lane >> 4;
  int n = n0 + col;
#pragma unroll
  for (int r = 0; r < 8; ++r) {
    int o = ot * 16 + r + 8 * grp;
    float val = d[r] + fc_b[o];
    if (MODE == 0) {
      val = val / (1.f + expf(-val));                     // SiLU
      out_t[((size_t)b * N_ + n) * I_ + o] = (_Float16)val;
    } else {
      val = tanhf(val) * lv_t[((size_t)b * N_ + n) * I_ + o];
      out_t[((size_t)b * N_ + n) * C_ + o] = (_Float16)val;  // y ch 0..63
    }
  }
}

// ---------------------------------------------------------------------------
// K8: attention pass 1: one block per (b, 16-query row-block).
// Score strip 16x4096 f32 = 256 KB in dynamic LDS (CDNA5 320KB/WG).
// Emits rowmax, 1/rowdenom, and deterministic partial column-sums.
// ---------------------------------------------------------------------------
__global__ __launch_bounds__(256) void k_att1(const _Float16* __restrict__ qc_t,
                                              const _Float16* __restrict__ k_t,
                                              float* __restrict__ rm_g,
                                              float* __restrict__ ird_g,
                                              float* __restrict__ pcs) {
  extern __shared__ float e_s[];                 // [16][4096]
  __shared__ float red[256], rm_s[16], rd_s[16];
  int b = blockIdx.x >> 8, rt = blockIdx.x & 255;
  int n0 = rt * 16;
  int lane = threadIdx.x & 31, w = threadIdx.x >> 5;
  const _Float16* qb = qc_t + ((size_t)b * N_) * I_;
  const _Float16* kb = k_t + ((size_t)b * N_) * I_;
  // A fragments (16 queries, K=64 in 2 steps), token-major -> b128 loads.
  v16h a0 = frag_a_h(qb + (size_t)n0 * I_, I_, 1, lane);
  v16h a1 = frag_a_h(qb + (size_t)n0 * I_ + 32, I_, 1, lane);
  int col = lane & 15, grp = lane >> 4;
  for (int mt = w; mt < NT_; mt += 8) {
    int m0 = mt * 16;
    __builtin_prefetch(kb + (size_t)(m0 + 128) * I_, 0, 3);
    v8f e = {};
    v16h b0 = frag_b_h(kb + (size_t)m0 * I_, 1, I_, lane);
    e = wmma16(a0, b0, e);
    v16h b1 = frag_b_h(kb + (size_t)m0 * I_ + 32, 1, I_, lane);
    e = wmma16(a1, b1, e);
#pragma unroll
    for (int r = 0; r < 8; ++r)
      e_s[(r + 8 * grp) * N_ + m0 + col] = e[r];
  }
  __syncthreads();
  int row = threadIdx.x >> 4, sub = threadIdx.x & 15;
  float mx = -3.4e38f;
  for (int c = sub; c < N_; c += 16) mx = fmaxf(mx, e_s[row * N_ + c]);
  red[threadIdx.x] = mx;
  __syncthreads();
  if (sub == 0) {
    float m2 = red[threadIdx.x];
    for (int j = 1; j < 16; ++j) m2 = fmaxf(m2, red[threadIdx.x + j]);
    rm_s[row] = m2;
  }
  __syncthreads();
  float m2 = rm_s[row];
  float s = 0.f;
  for (int c = sub; c < N_; c += 16) s += expf(e_s[row * N_ + c] - m2);
  red[threadIdx.x] = s;
  __syncthreads();
  if (sub == 0) {
    float dsum = red[threadIdx.x];
    for (int j = 1; j < 16; ++j) dsum += red[threadIdx.x + j];
    float inv = 1.f / dsum;
    rd_s[row] = inv;
    rm_g[b * N_ + n0 + row]  = rm_s[row];
    ird_g[b * N_ + n0 + row] = inv;
  }
  __syncthreads();
  for (int c = threadIdx.x; c < N_; c += 256) {
    float acc = 0.f;
#pragma unroll
    for (int r2 = 0; r2 < 16; ++r2)
      acc += expf(e_s[r2 * N_ + c] - rm_s[r2]) * rd_s[r2];
    pcs[((size_t)(b * NT_ + rt)) * N_ + c] = acc;
  }
}

// K9: reduce partial colsums -> 1/(1e-9 + colsum) (deterministic order).
__global__ __launch_bounds__(256) void k_colsum(const float* __restrict__ pcs,
                                                float* __restrict__ ci) {
  int idx = blockIdx.x * 256 + threadIdx.x;   // B*N total
  int b = idx / N_, m = idx % N_;
  float s = 0.f;
  for (int i = 0; i < NT_; ++i) s += pcs[((size_t)(b * NT_ + i)) * N_ + m];
  ci[idx] = 1.f / (1e-9f + s);
}

// ---------------------------------------------------------------------------
// K10: attention pass 2: one block (4 waves) per (b, 16-column tile).
// Recompute score tiles, exp with saved row stats, stage p in LDS as [m][n]
// (contiguous lane writes AND contiguous B-fragment reads), WMMA-accumulate
// v*p with f32 accumulators, divide by colsum -> y_t channels 64..127.
// ---------------------------------------------------------------------------
__global__ __launch_bounds__(128) void k_att2(const _Float16* __restrict__ qc_t,
                                              const _Float16* __restrict__ k_t,
                                              const _Float16* __restrict__ v_cm,
                                              const float* __restrict__ rm_g,
                                              const float* __restrict__ ird_g,
                                              const float* __restrict__ ci,
                                              _Float16* __restrict__ y_t) {
  __shared__ _Float16 p_s[16 * 64];       // [m_local][n_local]
  __shared__ float rm_s[N_], ird_s[N_];   // 32 KB row stats
  int b = blockIdx.x >> 8, mt = blockIdx.x & 255;
  int m0 = mt * 16;
  int lane = threadIdx.x & 31, w = threadIdx.x >> 5;
  for (int i = threadIdx.x; i < N_; i += 128) {
    rm_s[i]  = rm_g[b * N_ + i];
    ird_s[i] = ird_g[b * N_ + i];
  }
  __syncthreads();
  const _Float16* qb = qc_t + ((size_t)b * N_) * I_;
  const _Float16* kb = k_t + ((size_t)b * N_) * I_;
  const _Float16* vb = v_cm + (size_t)b * I_ * N_;
  // B fragments for the score tile are loop-invariant (same 16 cols): hoist.
  v16h kb0 = frag_b_h(kb + (size_t)m0 * I_, 1, I_, lane);
  v16h kb1 = frag_b_h(kb + (size_t)m0 * I_ + 32, 1, I_, lane);
  int col = lane & 15, grp = lane >> 4;
  int i0 = w * 16;                        // this wave's 16 v-rows
  v8f acc = {};
  for (int nc = 0; nc < 64; ++nc) {
    int nb = nc * 64;
    int n0q = nb + w * 16;                // this wave's 16 query rows
    __builtin_prefetch(qb + (size_t)(n0q + 64) * I_, 0, 3);
    v8f e = {};
    v16h a0 = frag_a_h(qb + (size_t)n0q * I_, I_, 1, lane);
    e = wmma16(a0, kb0, e);
    v16h a1 = frag_a_h(qb + (size_t)n0q * I_ + 32, I_, 1, lane);
    e = wmma16(a1, kb1, e);
#pragma unroll
    for (int r = 0; r < 8; ++r) {         // 8 consecutive halves per lane
      int nl = w * 16 + 8 * grp + r;
      int q = nb + nl;
      float pv = expf(e[r] - rm_s[q]) * ird_s[q];
      p_s[col * 64 + nl] = (_Float16)pv;
    }
    __syncthreads();
    v16h va0 = frag_a_h(vb + (size_t)i0 * N_ + nb, N_, 1, lane);
    v16h pb0 = frag_b_h(p_s, 1, 64, lane);
    acc = wmma16(va0, pb0, acc);
    v16h va1 = frag_a_h(vb + (size_t)i0 * N_ + nb + 32, N_, 1, lane);
    v16h pb1 = frag_b_h(p_s + 32, 1, 64, lane);
    acc = wmma16(va1, pb1, acc);
    __syncthreads();
  }
  float cinv = ci[b * N_ + m0 + col];
#pragma unroll
  for (int r = 0; r < 8; ++r) {
    int ch = 64 + i0 + r + 8 * grp;       // y channels 64..127, contiguous
    y_t[((size_t)b * N_ + m0 + col) * C_ + ch] = (_Float16)(acc[r] * cinv);
  }
}

// K13: tc GEMM [128x128] x [128xN] + bias -> y_fT (token-major f32).
__global__ __launch_bounds__(256) void k_tc(const _Float16* __restrict__ tc_wh,
                                            const float* __restrict__ tc_b,
                                            const _Float16* __restrict__ y_t,
                                            float* __restrict__ y_fT) {
  int ng = blockIdx.x & 31;
  int t  = blockIdx.x >> 5;
  int ot = t & 7, b = t >> 3;
  int lane = threadIdx.x & 31, w = threadIdx.x >> 5;
  int n0 = (ng * 8 + w) * 16;
  const _Float16* yb = y_t + ((size_t)b * N_) * C_;
  v8f d = {};
#pragma unroll
  for (int kk = 0; kk < 4; ++kk) {
    v16h a  = frag_a_h(tc_wh + (size_t)ot * 16 * C_ + kk * 32, C_, 1, lane);
    v16h bb = frag_b_h(yb + (size_t)n0 * C_ + kk * 32, 1, C_, lane);
    d = wmma16(a, bb, d);
  }
  int col = lane & 15, grp = lane >> 4;
  int n = n0 + col;
#pragma unroll
  for (int r = 0; r < 8; ++r) {
    int o = ot * 16 + r + 8 * grp;
    y_fT[((size_t)b * N_ + n) * C_ + o] = d[r] + tc_b[o];   // contiguous
  }
}

// K14: out = x + relu(gn2(y)); out is [b][c][n], y_fT is [b][n][c].
__global__ __launch_bounds__(256) void k_final(const float* __restrict__ x,
                                               const float* __restrict__ y_fT,
                                               const float* __restrict__ scale2,
                                               const float* __restrict__ shift2,
                                               float* __restrict__ out) {
  size_t idx = (size_t)blockIdx.x * 256 + threadIdx.x;
  if (idx >= (size_t)B_ * CN_) return;
  int b = (int)(idx / CN_);
  int c = (int)((idx / N_) % C_);
  int n = (int)(idx % N_);
  float yv = y_fT[((size_t)b * N_ + n) * C_ + c];
  float yn = yv * scale2[b * C_ + c] + shift2[b * C_ + c];
  out[idx] = x[idx] + fmaxf(yn, 0.f);
}

// ---------------------------------------------------------------------------
extern "C" void kernel_launch(void* const* d_in, const int* in_sizes, int n_in,
                              void* d_out, int out_size, void* d_ws, size_t ws_size,
                              hipStream_t stream) {
  const float* x      = (const float*)d_in[0];
  const float* bn_g   = (const float*)d_in[1];
  const float* bn_b   = (const float*)d_in[2];
  const float* gq_w   = (const float*)d_in[3];
  const float* gq_b   = (const float*)d_in[4];
  const float* gkv_w  = (const float*)d_in[5];
  const float* gkv_b  = (const float*)d_in[6];
  const float* lqkv_w = (const float*)d_in[7];
  const float* lqkv_b = (const float*)d_in[8];
  const float* dw_w   = (const float*)d_in[9];
  const float* dw_b   = (const float*)d_in[10];
  const float* fc_w   = (const float*)d_in[11];
  const float* fc_b   = (const float*)d_in[12];
  const float* tc_w   = (const float*)d_in[13];
  const float* tc_b   = (const float*)d_in[14];
  const float* an_g   = (const float*)d_in[15];
  const float* an_b   = (const float*)d_in[16];
  float* out = (float*)d_out;

  char* wsb = (char*)d_ws;
  size_t off = 0;
  auto take = [&](size_t bytes) -> char* {
    char* p = wsb + off;
    off = (off + bytes + 255) & ~(size_t)255;
    return p;
  };
  const size_t HIN = (size_t)B_ * I_ * N_;   // 1M elems
  float*    stat1   = (float*)take(B_ * SEGS * 2 * sizeof(float));
  float*    scale1  = (float*)take(B_ * C_ * sizeof(float));
  float*    shift1  = (float*)take(B_ * C_ * sizeof(float));
  _Float16* gq_wh   = (_Float16*)take((size_t)I_ * C_ * 2);
  _Float16* gkv_wh  = (_Float16*)take((size_t)2 * I_ * C_ * 2);
  _Float16* lqkv_wh = (_Float16*)take((size_t)3 * I_ * C_ * 2);
  _Float16* fc_wh   = (_Float16*)take((size_t)I_ * I_ * 2);
  _Float16* tc_wh   = (_Float16*)take((size_t)C_ * C_ * 2);
  _Float16* xn_t    = (_Float16*)take((size_t)B_ * CN_ * 2);
  _Float16* qc_t    = (_Float16*)take(HIN * 2);
  _Float16* k_t     = (_Float16*)take(HIN * 2);
  _Float16* v_cm    = (_Float16*)take(HIN * 2);
  _Float16* lq_t    = (_Float16*)take(HIN * 2);
  _Float16* lk_t    = (_Float16*)take(HIN * 2);
  float*    lv_t    = (float*)take(HIN * 4);
  _Float16* a_t     = (_Float16*)take(HIN * 2);
  _Float16* s1_t    = (_Float16*)take(HIN * 2);
  float*    rm_g    = (float*)take((size_t)B_ * N_ * 4);
  float*    ird_g   = (float*)take((size_t)B_ * N_ * 4);
  float*    pcs     = (float*)take((size_t)B_ * NT_ * N_ * 4);  // 16 MB
  float*    ci      = (float*)take((size_t)B_ * N_ * 4);
  _Float16* y_t     = (_Float16*)take((size_t)B_ * CN_ * 2);
  float*    y_fT    = (float*)take((size_t)B_ * CN_ * 4);
  float*    stat2   = (float*)take(B_ * SEGS * 2 * sizeof(float));
  float*    scale2  = (float*)take(B_ * C_ * sizeof(float));
  float*    shift2  = (float*)take(B_ * C_ * sizeof(float));
  (void)ws_size; (void)in_sizes; (void)n_in; (void)out_size;

  // 0) One-shot f16 weight conversion (removes cvt from GEMM inner loops)
  k_w2h<<<(I_ * C_ + 255) / 256, 256, 0, stream>>>(gq_w, gq_wh, I_ * C_);
  k_w2h<<<(2 * I_ * C_ + 255) / 256, 256, 0, stream>>>(gkv_w, gkv_wh, 2 * I_ * C_);
  k_w2h<<<(3 * I_ * C_ + 255) / 256, 256, 0, stream>>>(lqkv_w, lqkv_wh, 3 * I_ * C_);
  k_w2h<<<(I_ * I_ + 255) / 256, 256, 0, stream>>>(fc_w, fc_wh, I_ * I_);
  k_w2h<<<(C_ * C_ + 255) / 256, 256, 0, stream>>>(tc_w, tc_wh, C_ * C_);
  // 1) GroupNorm-1 stats, then fused normalize+transpose to token-major f16
  k_stats<<<B_ * SEGS, 256, 0, stream>>>(x, stat1);
  k_gn_final<<<B_, C_, 0, stream>>>(stat1, bn_g, bn_b, scale1, shift1);
  k_norm_t<<<B_ * 512, 256, 0, stream>>>(x, scale1, shift1, xn_t);
  // 2) All projections via WMMA (token-major outputs; v channel-major)
  k_project<<<B_ * 24 * 32, 256, 0, stream>>>(xn_t,
      gq_wh, gq_b, gkv_wh, gkv_b, lqkv_wh, lqkv_b, dw_w, dw_b,
      qc_t, k_t, v_cm, lq_t, lk_t, lv_t);
  // 3) Local branch: a = lq*lk ; fc+SiLU ; fc+tanh*lv -> y channels 0..63
  k_amul<<<(int)(HIN / 256), 256, 0, stream>>>(lq_t, lk_t, a_t);
  k_fc<0><<<B_ * 4 * 32, 256, 0, stream>>>(fc_wh, fc_b, a_t, nullptr, s1_t);
  k_fc<1><<<B_ * 4 * 32, 256, 0, stream>>>(fc_wh, fc_b, s1_t, lv_t, y_t);
  // 4) Attention: pass1 (256KB LDS score strip, row stats + partial colsums)
  k_att1<<<B_ * NT_, 256, 16 * N_ * sizeof(float), stream>>>(
      qc_t, k_t, rm_g, ird_g, pcs);
  k_colsum<<<(B_ * N_) / 256, 256, 0, stream>>>(pcs, ci);
  //    pass2 (streaming recompute + v*p WMMA) -> y channels 64..127
  k_att2<<<B_ * NT_, 128, 0, stream>>>(qc_t, k_t, v_cm, rm_g, ird_g, ci, y_t);
  // 5) tc GEMM, GroupNorm-2, residual + relu
  k_tc<<<B_ * 8 * 32, 256, 0, stream>>>(tc_wh, tc_b, y_t, y_fT);
  k_stats<<<B_ * SEGS, 256, 0, stream>>>(y_fT, stat2);
  k_gn_final<<<B_, C_, 0, stream>>>(stat2, an_g, an_b, scale2, shift2);
  k_final<<<(int)(((size_t)B_ * CN_) / 256), 256, 0, stream>>>(
      x, y_fT, scale2, shift2, out);
}
